// PointConv_27539330302431
// MI455X (gfx1250) — matrix operand order repaired and checked by reading, not compiled, not run on hardware
//
#include <hip/hip_runtime.h>
#include <cstdint>
#include <cstddef>

// Problem constants (match reference)
#define B_    16
#define N_    2048
#define D_    3
#define CIN_  64
#define COUT_ 64
#define NBHD_ 32
#define MID_  32
#define K_    16
#define BIG_  1e8f

typedef __attribute__((ext_vector_type(16))) _Float16 v16h;
typedef __attribute__((ext_vector_type(8)))  _Float16 v8h;
typedef __attribute__((ext_vector_type(8)))  float    v8f;

__device__ __forceinline__ float swish_f(float x) {
    return x / (1.0f + __expf(-x));
}

__device__ __forceinline__ v16h join8(v8h lo, v8h hi) {
    return __builtin_shufflevector(lo, hi, 0, 1, 2, 3, 4, 5, 6, 7,
                                           8, 9, 10, 11, 12, 13, 14, 15);
}

// ---------------------------------------------------------------------------
// Kernel 1: brute-force KNN. One wave32 per query point. Distances staged in
// LDS (2048 f32 per wave), then 32 rounds of argmin extraction with wave32
// shuffle reduction. Indices written to workspace.
// ---------------------------------------------------------------------------
__global__ void __launch_bounds__(256)
knn_kernel(const float* __restrict__ coords,
           const uint8_t* __restrict__ mask,
           int* __restrict__ knn)
{
    extern __shared__ float distS[];   // [8 waves][N_]
    const int wave = threadIdx.x >> 5;
    const int lane = threadIdx.x & 31;
    const int q    = blockIdx.x * 8 + wave;       // query id in [0, B*N)
    const int b    = q / N_;
    const int m    = q % N_;
    float* dist = distS + wave * N_;

    const float qx = coords[((size_t)(b * N_ + m)) * 3 + 0];
    const float qy = coords[((size_t)(b * N_ + m)) * 3 + 1];
    const float qz = coords[((size_t)(b * N_ + m)) * 3 + 2];

    for (int i = 0; i < 64; ++i) {
        const int n = lane + (i << 5);
        const float* c = coords + ((size_t)(b * N_ + n)) * 3;
        const float dx = qx - c[0];
        const float dy = qy - c[1];
        const float dz = qz - c[2];
        float d = dx * dx + dy * dy + dz * dz;
        if (!mask[b * N_ + n]) d = BIG_;
        dist[n] = d;
    }
    __syncthreads();

    for (int r = 0; r < NBHD_; ++r) {
        float bv = 4.0f * BIG_;
        int   bi = 0x7fffffff;
        for (int i = 0; i < 64; ++i) {
            const int n = lane + (i << 5);
            const float d = dist[n];
            if (d < bv) { bv = d; bi = n; }
        }
        #pragma unroll
        for (int off = 16; off > 0; off >>= 1) {
            const float ov = __shfl_xor(bv, off, 32);
            const int   oi = __shfl_xor(bi, off, 32);
            if (ov < bv || (ov == bv && oi < bi)) { bv = ov; bi = oi; }
        }
        if (lane == 0) {
            knn[(size_t)q * NBHD_ + r] = bi;
            dist[bi] = 4.0f * BIG_;
        }
        __syncthreads();
    }
}

// ---------------------------------------------------------------------------
// Kernel 1b: pre-pack wl into f16 WMMA B-fragment order with the flat-K axis
// PERMUTED to f = k_out*64 + c (matching pcS's write-friendly layout):
//   original row of flat index f is (f&63)*16 + (f>>6)   [= c*K + k_out]
//   wlPk[((t*32+ks)*32 + lane)*16 + h] covers new-flat rows
//       f = 32*ks + (lane>=16 ? 16:0) + h,  column n = 16*t + (lane&15)
// so stage-3 B fragments are two contiguous 16B global loads per lane.
// ---------------------------------------------------------------------------
__global__ void __launch_bounds__(256)
pack_wl_kernel(const float* __restrict__ wl, _Float16* __restrict__ wlPk)
{
    const int i = blockIdx.x * 256 + threadIdx.x;   // [0, 4*32*32*16)
    const int h    = i & 15;
    const int lane = (i >> 4) & 31;
    const int ks   = (i >> 9) & 31;
    const int t    = i >> 14;
    const int f    = 32 * ks + ((lane & 16) ? 16 : 0) + h;  // permuted flat-K
    const int orig = (f & 63) * K_ + (f >> 6);              // c*K + k_out
    const int n    = 16 * t + (lane & 15);
    wlPk[i] = (_Float16)wl[orig * COUT_ + n];
}

// ---------------------------------------------------------------------------
// Kernel 2: fused gather + WeightNet MLP + WMMA aggregation + WMMA final GEMM.
// One block = 16 query points.
//   LDS layout (dynamic, ~114 KB), all operands stored K-contiguous so every
//   WMMA fragment is two ds_load_b128 and the pc writeback is one
//   ds_store_b128 per tile:
//     idxS : 16*32 int                        (2 KB)
//     vT   : [16 pt][64 ch][32 nb] f16        (64 KB)   (A of stage 2)
//     wT   : [16 pt][16 k][32 nb]  f16        (16 KB)   (B of stage 2)
//     pcS  : [16 pt][k*64+c]       f16        (32 KB)   (A of stage 3,
//                                                        permuted flat-K)
// ---------------------------------------------------------------------------
__global__ void __launch_bounds__(256)
pointconv_kernel(const float* __restrict__ coords,
                 const float* __restrict__ values,
                 const uint8_t* __restrict__ mask,
                 const float* __restrict__ w1, const float* __restrict__ b1,
                 const float* __restrict__ w2, const float* __restrict__ b2,
                 const float* __restrict__ w3, const float* __restrict__ b3,
                 const float* __restrict__ bl,
                 const int* __restrict__ knn,
                 const _Float16* __restrict__ wlPk,
                 float* __restrict__ out)
{
    extern __shared__ char smem[];
    int*      idxS = (int*)smem;                                   // 2048 B
    _Float16* vT   = (_Float16*)(smem + 2048);                     // 65536 B
    _Float16* wT   = (_Float16*)(smem + 2048 + 65536);             // 16384 B
    _Float16* pcS  = (_Float16*)(smem + 2048 + 65536 + 16384);     // 32768 B

    const int tid  = threadIdx.x;
    const int lane = tid & 31;
    const int wave = tid >> 5;
    const int b    = (blockIdx.x * 16) / N_;
    const int m0   = (blockIdx.x * 16) % N_;
    const size_t qbase = (size_t)b * N_ + m0;

    const int laneHi = (lane & 16) ? 1 : 0;   // lane in [16,32)
    const int lm     = lane & 15;

    // --- load neighbor indices ---
    for (int t = tid; t < 16 * NBHD_; t += 256)
        idxS[t] = knn[qbase * NBHD_ + t];
    __syncthreads();

    // --- stage 1: gather values (f32->f16, transposed) + WeightNet MLP ---
    #pragma unroll 1
    for (int t = tid; t < 16 * NBHD_; t += 256) {
        const int pt = t >> 5;
        const int nb = t & 31;
        const int j  = idxS[t];
        const float msk = mask[b * N_ + j] ? 1.0f : 0.0f;

        // gather 64 channels; write transposed: vT[pt][c][nb]
        const float4* vr = (const float4*)(values + ((size_t)(b * N_ + j)) * CIN_);
        _Float16* vd = vT + pt * CIN_ * NBHD_ + nb;
        #pragma unroll
        for (int c4 = 0; c4 < 16; ++c4) {
            const float4 v = vr[c4];
            vd[(c4 * 4 + 0) * NBHD_] = (_Float16)(v.x * msk);
            vd[(c4 * 4 + 1) * NBHD_] = (_Float16)(v.y * msk);
            vd[(c4 * 4 + 2) * NBHD_] = (_Float16)(v.z * msk);
            vd[(c4 * 4 + 3) * NBHD_] = (_Float16)(v.w * msk);
        }

        // relative coordinates
        const float* qc = coords + (qbase + pt) * 3;
        const float* nc = coords + ((size_t)(b * N_ + j)) * 3;
        const float dx = qc[0] - nc[0];
        const float dy = qc[1] - nc[1];
        const float dz = qc[2] - nc[2];

        // WeightNet: 3 -> 32 -> 32 -> 16, swish (weights are wave-uniform
        // -> scalar loads; h1/h2 stay register-resident via full unroll)
        float h1[MID_], h2[MID_];
        #pragma unroll
        for (int i = 0; i < MID_; ++i)
            h1[i] = swish_f(b1[i] + dx * w1[0 * MID_ + i]
                                  + dy * w1[1 * MID_ + i]
                                  + dz * w1[2 * MID_ + i]);
        #pragma unroll
        for (int i = 0; i < MID_; ++i) {
            float s = b2[i];
            #pragma unroll
            for (int jx = 0; jx < MID_; ++jx) s += h1[jx] * w2[jx * MID_ + i];
            h2[i] = swish_f(s);
        }
        _Float16* wd = wT + pt * K_ * NBHD_ + nb;   // wT[pt][k][nb]
        #pragma unroll
        for (int k = 0; k < K_; ++k) {
            float s = b3[k];
            #pragma unroll
            for (int jx = 0; jx < MID_; ++jx) s += h2[jx] * w3[jx * K_ + k];
            wd[k * NBHD_] = (_Float16)(swish_f(s) * msk);
        }
    }
    __syncthreads();

    // --- stage 2: pc[c,k] = sum_n v[n,c] * w[n,k] per point ---
    // (64x32)@(32x16) = 4 tiles of v_wmma_f32_16x16x32_f16.
    for (int pt = wave; pt < 16; pt += 8) {
        const _Float16* vp = vT + pt * CIN_ * NBHD_;
        const _Float16* wp = wT + pt * K_ * NBHD_;

        // B fragment: lane holds column k_out=lm, K-rows n = laneHi*8 + {0..7, 16..23}
        const _Float16* wrow = wp + lm * NBHD_ + laneHi * 8;
        const v16h bfrag = join8(*(const v8h*)(wrow), *(const v8h*)(wrow + 16));

        #pragma unroll
        for (int t = 0; t < 4; ++t) {
            // A fragment: lane holds row c = 16t+lm; K chunks per ISA layout:
            //   lane<16: n in {0..7, 16..23};  lane>=16: n in {8..15, 24..31}
            const _Float16* arow = vp + (t * 16 + lm) * NBHD_ + laneHi * 8;
            const v16h afrag = join8(*(const v8h*)(arow),
                                     *(const v8h*)(arow + 16));
            v8f acc = {};
            acc = __builtin_amdgcn_wmma_f32_16x16x32_f16(
                      false, afrag, false, bfrag, (short)0, acc, false, false);
            // D layout: reg r -> (c = 16t + 8*laneHi + r, k = lm).
            // pcS flat index = k*64 + c  ->  lane's 8 values are contiguous:
            // one aligned 16B store.
            v8h pk;
            #pragma unroll
            for (int r = 0; r < 8; ++r) pk[r] = (_Float16)acc[r];
            *(v8h*)(pcS + pt * 1024 + lm * 64 + t * 16 + laneHi * 8) = pk;
        }
    }
    __syncthreads();

    // --- stage 3: out(16x64) = pc(16x1024) @ wl(1024x64) + bl ---
    // waves 0..3 each own a 16-column tile; 32 chained WMMA over the
    // permuted flat-K = 1024 axis (wlPk rows permuted identically).
    if (wave < 4) {
        v8f acc = {};
        const _Float16* wlbase = wlPk + ((size_t)wave * 32 * 32 + lane) * 16;
        for (int ks = 0; ks < 32; ++ks) {
            const int kbase = ks * 32;
            // A from pcS: row m = point = lm, flat-k contiguous chunks
            const _Float16* arow = pcS + lm * 1024 + kbase + laneHi * 16;
            const v16h afrag = join8(*(const v8h*)(arow),
                                     *(const v8h*)(arow + 8));
            // B from pre-packed wl: one 32B contiguous run per lane
            const _Float16* brow = wlbase + (size_t)ks * 32 * 16;
            const v16h bfrag = join8(*(const v8h*)(brow),
                                     *(const v8h*)(brow + 8));
            // prefetch next k-step's chunk (global_prefetch_b8)
            if (ks < 31) __builtin_prefetch(brow + 32 * 16, 0, 1);
            acc = __builtin_amdgcn_wmma_f32_16x16x32_f16(
                      false, afrag, false, bfrag, (short)0, acc, false, false);
        }
        #pragma unroll
        for (int r = 0; r < 8; ++r) {
            const int p   = r + laneHi * 8;        // point within block
            const int col = wave * 16 + lm;
            float v = acc[r] + bl[col];
            if (!mask[qbase + p]) v = 0.0f;
            out[(qbase + p) * COUT_ + col] = v;
        }
    }
}

// ---------------------------------------------------------------------------
// Kernel 3: tuple outputs 1 and 3 — copy coords, write mask as float.
// ---------------------------------------------------------------------------
__global__ void copy_coords_mask(const float* __restrict__ coords,
                                 const uint8_t* __restrict__ mask,
                                 float* __restrict__ outc,
                                 float* __restrict__ outm)
{
    const int i = blockIdx.x * blockDim.x + threadIdx.x;
    if (i < B_ * N_ * D_) outc[i] = coords[i];
    if (i < B_ * N_)      outm[i] = mask[i] ? 1.0f : 0.0f;
}

// ---------------------------------------------------------------------------
extern "C" void kernel_launch(void* const* d_in, const int* in_sizes, int n_in,
                              void* d_out, int out_size, void* d_ws, size_t ws_size,
                              hipStream_t stream)
{
    const float*   coords = (const float*)d_in[0];
    const float*   values = (const float*)d_in[1];
    const uint8_t* mask   = (const uint8_t*)d_in[2];
    const float* w1 = (const float*)d_in[3];
    const float* b1 = (const float*)d_in[4];
    const float* w2 = (const float*)d_in[5];
    const float* b2 = (const float*)d_in[6];
    const float* w3 = (const float*)d_in[7];
    const float* b3 = (const float*)d_in[8];
    const float* wl = (const float*)d_in[9];
    const float* bl = (const float*)d_in[10];

    // workspace: [0, 4MB) knn indices, [4MB, 4MB+128KB) packed wl (f16)
    int*      knn  = (int*)d_ws;
    _Float16* wlPk = (_Float16*)((char*)d_ws + (size_t)B_ * N_ * NBHD_ * 4);

    float* outc = (float*)d_out;                          // (B,N,D)
    float* outv = outc + (size_t)B_ * N_ * D_;            // (B,N,COUT)
    float* outm = outv + (size_t)B_ * N_ * COUT_;         // (B,N)

    const int smemA = 8 * N_ * (int)sizeof(float);                 // 64 KB
    const int smemB = 2048 + 65536 + 16384 + 32768;                // ~114 KB

    (void)hipFuncSetAttribute((const void*)knn_kernel,
                              hipFuncAttributeMaxDynamicSharedMemorySize, smemA);
    (void)hipFuncSetAttribute((const void*)pointconv_kernel,
                              hipFuncAttributeMaxDynamicSharedMemorySize, smemB);

    pack_wl_kernel<<<(4 * 32 * 32 * 16) / 256, 256, 0, stream>>>(wl, wlPk);
    knn_kernel<<<(B_ * N_) / 8, 256, smemA, stream>>>(coords, mask, knn);
    pointconv_kernel<<<(B_ * N_) / 16, 256, smemB, stream>>>(
        coords, values, mask, w1, b1, w2, b2, w3, b3, bl, knn, wlPk, outv);
    copy_coords_mask<<<(B_ * N_ * D_ + 255) / 256, 256, 0, stream>>>(
        coords, mask, outc, outm);
}
